// FixedGraphAttentionLayer_19267223290029
// MI455X (gfx1250) — matrix operand order repaired and checked by reading, not compiled, not run on hardware
//
#include <hip/hip_runtime.h>

// Fused GAT-style layer for MI455X (gfx1250, wave32, WMMA).
//   h = x(16x128) @ W(128x64)   -> v_wmma_f32_16x16x32_bf16, fp32 accum
//   scores/softmax/aggregate/ELU fused in registers (cross-lane shuffles).
// Memory-bound: 655 MB of x @ 23.3 TB/s ~ 28us; WMMA keeps compute off the
// critical path (fp32 VALU could not).

typedef __attribute__((ext_vector_type(16))) __bf16 v16bf;
typedef __attribute__((ext_vector_type(8)))  float  v8f;

#define ALPHA 0.2f
#define IN_F  128
#define OUT_F 64
#define DNEI  16   // neighborhood size (softmax axis)

__global__ __launch_bounds__(256) void gat_fused_kernel(
    const float* __restrict__ x, const float* __restrict__ W,
    const float* __restrict__ a, float* __restrict__ out, int numTiles)
{
  const int lane = threadIdx.x & 31;
  const int wave = threadIdx.x >> 5;
  const int half = lane >> 4;    // which 16-lane half of the wave
  const int col  = lane & 15;    // N column (B/C/D) and A row for this lane

  // ---- W (128x64 f32) -> bf16 WMMA B-fragments, register resident -------
  // B layout (16-bit, 32x16), matching A: element e <-> K = 8*half + e (e<8),
  // K = 16 + 8*half + (e-8) (e>=8); column N = lane&15.
  v16bf Bf[4][4];
#pragma unroll
  for (int ks = 0; ks < 4; ++ks) {
#pragma unroll
    for (int nt = 0; nt < 4; ++nt) {
#pragma unroll
      for (int e = 0; e < 16; ++e) {
        const int k = ks * 32 + 8 * half + (e < 8 ? e : e + 8);
        Bf[ks][nt][e] = (__bf16)W[k * OUT_F + nt * 16 + col];
      }
    }
  }

  float a_nb[4], a_ct[4];
#pragma unroll
  for (int nt = 0; nt < 4; ++nt) {
    a_nb[nt] = a[nt * 16 + col];          // a[:64]  for neighbor score
    a_ct[nt] = a[OUT_F + nt * 16 + col];  // a[64:]  for center score
  }

  const int wavesPerBlock = blockDim.x >> 5;
  const int stride = gridDim.x * wavesPerBlock;

  for (int tile = blockIdx.x * wavesPerBlock + wave; tile < numTiles; tile += stride) {
    const float* xt = x + (size_t)tile * (DNEI * IN_F);

    // Prefetch this wave's next tile (32 lanes x 256B = full 8KB tile).
    __builtin_prefetch(xt + (size_t)stride * (DNEI * IN_F) + lane * 64, 0, 1);

    // ---- A fragments (16-bit A 16x32 layout) ---------------------------
    // lane<16: elements 0..7 = K 0..7,  8..15 = K 16..23 (of row col)
    // lane>=16: shifted by +8 in K.
    v16bf Af[4];
#pragma unroll
    for (int ks = 0; ks < 4; ++ks) {
      const float* xr = xt + col * IN_F + ks * 32 + 8 * half;
      const float4 q0 = *(const float4*)(xr + 0);
      const float4 q1 = *(const float4*)(xr + 4);
      const float4 q2 = *(const float4*)(xr + 16);
      const float4 q3 = *(const float4*)(xr + 20);
      const float t[16] = {q0.x, q0.y, q0.z, q0.w, q1.x, q1.y, q1.z, q1.w,
                           q2.x, q2.y, q2.z, q2.w, q3.x, q3.y, q3.z, q3.w};
#pragma unroll
      for (int e = 0; e < 16; ++e) Af[ks][e] = (__bf16)t[e];
    }

    // ---- h = A @ B : 4 N-tiles x 4 K-steps of WMMA ---------------------
    // C/D layout: VGPR r holds h[d = r + 8*half][o = nt*16 + col].
    v8f h[4];
#pragma unroll
    for (int nt = 0; nt < 4; ++nt) {
      v8f c = {};
#pragma unroll
      for (int ks = 0; ks < 4; ++ks)
        c = __builtin_amdgcn_wmma_f32_16x16x32_bf16(
            /*neg_a=*/false, Af[ks], /*neg_b=*/false, Bf[ks][nt],
            /*c_mod=*/(short)0, c, /*reuse_a=*/false, /*reuse_b=*/false);
      h[nt] = c;
    }

    // ---- scores: s_nb[d] = h[d,:] . a[:64] (reduce over 16 lanes/half) --
    float sc[8];
#pragma unroll
    for (int r = 0; r < 8; ++r) {
      float p = h[0][r] * a_nb[0] + h[1][r] * a_nb[1] +
                h[2][r] * a_nb[2] + h[3][r] * a_nb[3];
      p += __shfl_xor(p, 1, 32);
      p += __shfl_xor(p, 2, 32);
      p += __shfl_xor(p, 4, 32);
      p += __shfl_xor(p, 8, 32);   // sum within the 16-lane half
      sc[r] = p;                   // s_nb for d = r + 8*half
    }
    // s_ctr = h[0,:] . a[64:]  (d=0 lives in half 0, r=0)
    float pc = h[0][0] * a_ct[0] + h[1][0] * a_ct[1] +
               h[2][0] * a_ct[2] + h[3][0] * a_ct[3];
    pc += __shfl_xor(pc, 1, 32);
    pc += __shfl_xor(pc, 2, 32);
    pc += __shfl_xor(pc, 4, 32);
    pc += __shfl_xor(pc, 8, 32);
    pc = __shfl(pc, col, 32);      // broadcast half-0 result to both halves

    // leaky_relu(s_nb + s_ctr)
#pragma unroll
    for (int r = 0; r < 8; ++r) {
      const float s = sc[r] + pc;
      sc[r] = (s >= 0.f) ? s : ALPHA * s;
    }

    // ---- softmax over the 16 neighbors (8 in-lane + cross-half) --------
    float m = sc[0];
#pragma unroll
    for (int r = 1; r < 8; ++r) m = fmaxf(m, sc[r]);
    m = fmaxf(m, __shfl_xor(m, 16, 32));
    float sum = 0.f;
#pragma unroll
    for (int r = 0; r < 8; ++r) { sc[r] = __expf(sc[r] - m); sum += sc[r]; }
    sum += __shfl_xor(sum, 16, 32);
    const float inv = 1.0f / sum;

    // ---- out[o] = sum_d att[d] * h[d][o], then ELU ---------------------
    float o4[4];
#pragma unroll
    for (int nt = 0; nt < 4; ++nt) {
      float t = 0.f;
#pragma unroll
      for (int r = 0; r < 8; ++r) t += sc[r] * h[nt][r];
      t *= inv;
      t += __shfl_xor(t, 16, 32);              // add the other half's 8 d's
      o4[nt] = (t > 0.f) ? t : (__expf(t) - 1.f);
    }

    // Coalesced store: half 0 writes o[0..31], half 1 writes o[32..63].
    float* op = out + (size_t)tile * OUT_F;
    op[(2 * half + 0) * 16 + col] = o4[2 * half + 0];
    op[(2 * half + 1) * 16 + col] = o4[2 * half + 1];
  }
}

extern "C" void kernel_launch(void* const* d_in, const int* in_sizes, int n_in,
                              void* d_out, int out_size, void* d_ws, size_t ws_size,
                              hipStream_t stream) {
  const float* x = (const float*)d_in[0];
  const float* W = (const float*)d_in[1];
  const float* a = (const float*)d_in[2];
  float* out = (float*)d_out;

  const int numTiles = in_sizes[0] / (DNEI * IN_F);   // 80000 for the ref shapes
  const int wavesPerBlock = 8;                        // 256 threads = 8 wave32
  int blocks = (numTiles + wavesPerBlock - 1) / wavesPerBlock;
  if (blocks > 1024) blocks = 1024;                   // grid-stride, ~10 tiles/wave
  if (blocks < 1) blocks = 1;

  gat_fused_kernel<<<blocks, 256, 0, stream>>>(x, W, a, out, numTiles);
}
